// PolicyNetwork_32933809226462
// MI455X (gfx1250) — matrix operand order repaired and checked by reading, not compiled, not run on hardware
//
#include <hip/hip_runtime.h>
#include <hip/hip_bf16.h>
#include <cstdint>

typedef __attribute__((ext_vector_type(16))) _Float16 v16h;
typedef __attribute__((ext_vector_type(8)))  float    v8f;

#define DD   768
#define SS   2048
#define BB   8
#define BSR  (BB*SS)          // 16384 rows
#define PSTR 2056             // prob LDS stride (halves): dword stride 1028 %64 == 4 -> conflict-free
#define QSTR 776              // q LDS stride (halves):    dword stride 388  %64 == 4 -> conflict-free

union Frag16 { v16h h; _Float16 e[16]; unsigned u[8]; };

// A fragment (16x32, f16) from row-major [.., ld]; ISA layout (CDNA5 ISA):
// lane m=lane&15, hi=lane>>4; vgpr g<4: K=hi*8+2g+h ; g>=4: K=16+hi*8+2(g-4)+h
__device__ inline v16h load_a_frag(const _Float16* p, int ld, int rowBase, int kBase, int lane) {
    int m = lane & 15, hi = (lane >> 4) & 1;
    const _Float16* row = p + (size_t)(rowBase + m) * ld + kBase + hi * 8;
    Frag16 f;
#pragma unroll
    for (int g = 0; g < 4; ++g) f.u[g]     = *reinterpret_cast<const unsigned*>(row + 2 * g);
#pragma unroll
    for (int g = 0; g < 4; ++g) f.u[4 + g] = *reinterpret_cast<const unsigned*>(row + 16 + 2 * g);
    return f.h;
}

// B fragment (32x16, f16) where element (K,N) lives at p[N*ld + K]:
// lane n=lane&15, hi=lane>>4; vgpr g<4: K=hi*8+2g+h ; g>=4: K=16+hi*8+2(g-4)+h
__device__ inline v16h load_b_frag(const _Float16* p, int ld, int nBase, int kBase, int lane) {
    int n = lane & 15, hi = (lane >> 4) & 1;
    const _Float16* row = p + (size_t)(nBase + n) * ld + kBase + hi * 8;
    Frag16 f;
#pragma unroll
    for (int g = 0; g < 4; ++g) f.u[g]     = *reinterpret_cast<const unsigned*>(row + 2 * g);
#pragma unroll
    for (int g = 0; g < 4; ++g) f.u[4 + g] = *reinterpret_cast<const unsigned*>(row + 16 + 2 * g);
    return f.h;
}

__device__ inline v8f wmma_f16(v16h a, v16h b, v8f c) {
    return __builtin_amdgcn_wmma_f32_16x16x32_f16(false, a, false, b, (short)0, c, false, false);
}

// ---------------- kernel 1: weight transpose + f16 convert  Wt[n*D+k] = W[k*D+n]
__global__ void k_cvt_w(const float* __restrict__ W, _Float16* __restrict__ Wt) {
    int idx = blockIdx.x * 256 + threadIdx.x;
    if (idx >= DD * DD) return;
    int n = idx / DD, k = idx % DD;
    Wt[idx] = (_Float16)W[(size_t)k * DD + n];
}

// ---------------- kernel 2: per-batch valid lengths
__global__ void k_lengths(const int* __restrict__ mask, int* __restrict__ len) {
    __shared__ int red[256];
    int b = blockIdx.x, t = threadIdx.x, s = 0;
    for (int c = t; c < SS; c += 256) s += (mask[b * SS + c] != 0);
    red[t] = s; __syncthreads();
    for (int off = 128; off > 0; off >>= 1) { if (t < off) red[t] += red[t + off]; __syncthreads(); }
    if (t == 0) len[b] = red[0];
}

// ---------------- kernel 3: custom LayerNorm (ddof=1, eps=1e-6), emit f32 + f16
__global__ void k_norm(const float* __restrict__ x, const float* __restrict__ a2,
                       const float* __restrict__ b2, float* __restrict__ ctx,
                       _Float16* __restrict__ ctxh) {
    __shared__ float red[256];
    int r = blockIdx.x, t = threadIdx.x;
    const float* xr = x + (size_t)r * DD;
    float v0 = xr[t], v1 = xr[t + 256], v2 = xr[t + 512];
    red[t] = v0 + v1 + v2; __syncthreads();
    for (int off = 128; off > 0; off >>= 1) { if (t < off) red[t] += red[t + off]; __syncthreads(); }
    float mean = red[0] * (1.0f / (float)DD); __syncthreads();
    float d0 = v0 - mean, d1 = v1 - mean, d2 = v2 - mean;
    red[t] = d0 * d0 + d1 * d1 + d2 * d2; __syncthreads();
    for (int off = 128; off > 0; off >>= 1) { if (t < off) red[t] += red[t + off]; __syncthreads(); }
    float stddev = sqrtf(red[0] * (1.0f / (float)(DD - 1)));
    float sc = 1.0f / (stddev + 1e-6f);
    float y0 = a2[t] * d0 * sc + b2[t];
    float y1 = a2[t + 256] * d1 * sc + b2[t + 256];
    float y2 = a2[t + 512] * d2 * sc + b2[t + 512];
    size_t base = (size_t)r * DD;
    ctx[base + t] = y0; ctx[base + t + 256] = y1; ctx[base + t + 512] = y2;
    ctxh[base + t] = (_Float16)y0; ctxh[base + t + 256] = (_Float16)y1; ctxh[base + t + 512] = (_Float16)y2;
}

// ---------------- kernel 4: QKV GEMM; 2 M-tiles x 4 N-tiles per wave
// wave job id -> (32-row group, matrix, 64-col group); V written transposed per batch
__global__ void k_qkv(const _Float16* __restrict__ ctxh,
                      const _Float16* __restrict__ Wqt, const _Float16* __restrict__ Wkt,
                      const _Float16* __restrict__ Wvt,
                      const float* __restrict__ bq, const float* __restrict__ bk,
                      const float* __restrict__ bv,
                      _Float16* __restrict__ qh, _Float16* __restrict__ kh,
                      _Float16* __restrict__ vT) {
    int wid  = blockIdx.x * 8 + (threadIdx.x >> 5);   // 0..18431
    int lane = threadIdx.x & 31;
    int mg   = wid / 36;                              // 0..511 : 32-row group
    int job  = wid - mg * 36;                         // 0..35
    int mat  = job / 12;                              // 0=q 1=k 2=v
    int ntB  = (job - mat * 12) * 4;                  // N tile base (4 tiles = 64 cols)
    int rowBase = mg * 32;
    const _Float16* Wt = (mat == 0) ? Wqt : (mat == 1) ? Wkt : Wvt;
    const float* bias  = (mat == 0) ? bq  : (mat == 1) ? bk  : bv;

    v8f acc[2][4] = {};
    for (int kk = 0; kk < DD; kk += 32) {
        v16h a0 = load_a_frag(ctxh, DD, rowBase,      kk, lane);
        v16h a1 = load_a_frag(ctxh, DD, rowBase + 16, kk, lane);
#pragma unroll
        for (int j = 0; j < 4; ++j) {
            v16h b = load_b_frag(Wt, DD, (ntB + j) * 16, kk, lane);
            acc[0][j] = wmma_f16(a0, b, acc[0][j]);
            acc[1][j] = wmma_f16(a1, b, acc[1][j]);
        }
    }
    int n = lane & 15, hi = (lane >> 4) & 1;
    if (mat < 2) {
        _Float16* dst = (mat == 0) ? qh : kh;
#pragma unroll
        for (int j = 0; j < 4; ++j) {
            int col = (ntB + j) * 16 + n;
            float bbv = bias[col];
#pragma unroll
            for (int i = 0; i < 2; ++i)
#pragma unroll
                for (int r = 0; r < 8; ++r)
                    dst[(size_t)(rowBase + i * 16 + r + hi * 8) * DD + col] =
                        (_Float16)(acc[i][j][r] + bbv);
        }
    } else {
        int b_ = rowBase / SS;
        int lrow0 = rowBase - b_ * SS;
#pragma unroll
        for (int j = 0; j < 4; ++j) {
            int col = (ntB + j) * 16 + n;
            float bbv = bias[col];
            _Float16* dst = vT + (size_t)b_ * DD * SS + (size_t)col * SS;
#pragma unroll
            for (int i = 0; i < 2; ++i)
#pragma unroll
                for (int r = 0; r < 8; ++r)
                    dst[lrow0 + i * 16 + r + hi * 8] = (_Float16)(acc[i][j][r] + bbv);
        }
    }
}

// ---------------- kernel 5: attention; 16-row stripe per WG; f16 probs in LDS (~92KB)
__global__ __launch_bounds__(256)
void k_attn(const _Float16* __restrict__ qh, const _Float16* __restrict__ kh,
            const _Float16* __restrict__ vT, const int* __restrict__ mask,
            const int* __restrict__ len, _Float16* __restrict__ cl) {
    __shared__ _Float16 sP[16 * PSTR];     // masked scores -> exp values (f16)
    __shared__ _Float16 sQ[16 * QSTR];     // staged q tile
    __shared__ float    part[16][17];
    __shared__ float    rowmax[16], rinv[16];

    int tid = threadIdx.x, wave = tid >> 5, lane = tid & 31;
    int blk = blockIdx.x;                  // 0..1023
    int b = blk >> 7;                      // / (S/16)
    int rtl = (blk & 127) * 16;            // local row base within batch
    size_t rowG = (size_t)b * SS + rtl;

    for (int i = tid; i < 16 * DD; i += 256) {
        int m = i / DD, k = i - m * DD;
        sQ[m * QSTR + k] = qh[(rowG + m) * DD + k];
    }
    __syncthreads();
    int L = len[b];
    const _Float16 MASKV = (_Float16)(-30000.0f);

    // phase 1: scores = qk^T / D, masked; 2 key-tiles per A-fragment
    for (int jp = wave; jp < SS / 32; jp += 8) {
        int jt0 = jp * 2;
        v8f acc0 = {}, acc1 = {};
        for (int kk = 0; kk < DD; kk += 32) {
            v16h a  = load_a_frag(sQ, QSTR, 0, kk, lane);
            v16h b0 = load_b_frag(kh + (size_t)b * SS * DD, DD, jt0 * 16,      kk, lane);
            v16h b1 = load_b_frag(kh + (size_t)b * SS * DD, DD, jt0 * 16 + 16, kk, lane);
            acc0 = wmma_f16(a, b0, acc0);
            acc1 = wmma_f16(a, b1, acc1);
        }
        int n = lane & 15, hi = (lane >> 4) & 1;
        int c0 = jt0 * 16 + n, c1 = c0 + 16;
        int mk0 = mask[b * SS + c0], mk1 = mask[b * SS + c1];
#pragma unroll
        for (int r = 0; r < 8; ++r) {
            int mrow = r + hi * 8;
            int rl = rtl + mrow;
            bool ok0 = (mk0 != 0) && (rl < L) && (c0 <= rl);
            bool ok1 = (mk1 != 0) && (rl < L) && (c1 <= rl);
            sP[mrow * PSTR + c0] = ok0 ? (_Float16)(acc0[r] * (1.0f / 768.0f)) : MASKV;
            sP[mrow * PSTR + c1] = ok1 ? (_Float16)(acc1[r] * (1.0f / 768.0f)) : MASKV;
        }
    }
    __syncthreads();

    // softmax stats (f32 math on f16 scores): 16 threads per row
    int row = tid >> 4, sub = tid & 15;
    float mx = -3.4e38f;
    for (int c = sub; c < SS; c += 16) mx = fmaxf(mx, (float)sP[row * PSTR + c]);
    part[row][sub] = mx; __syncthreads();
    if (sub == 0) {
        float m2 = part[row][0];
        for (int i = 1; i < 16; ++i) m2 = fmaxf(m2, part[row][i]);
        rowmax[row] = m2;
    }
    __syncthreads();
    float m = rowmax[row];
    float sum = 0.0f;
    for (int c = sub; c < SS; c += 16) {
        float e = __expf((float)sP[row * PSTR + c] - m);
        sP[row * PSTR + c] = (_Float16)e; sum += e;
    }
    part[row][sub] = sum; __syncthreads();
    if (sub == 0) {
        float t = 0.0f;
        for (int i = 0; i < 16; ++i) t += part[row][i];
        rinv[row] = 1.0f / t;
    }
    __syncthreads();

    // phase 2: O = P @ V; 3 N-tiles per P-fragment (6 per wave)
    for (int ntg = 0; ntg < 2; ++ntg) {
        int nt0 = wave * 6 + ntg * 3;
        v8f acc[3] = {};
        for (int jt = 0; jt < SS / 32; ++jt) {
            v16h a = load_a_frag(sP, PSTR, 0, jt * 32, lane);
#pragma unroll
            for (int j = 0; j < 3; ++j) {
                v16h bf = load_b_frag(vT + (size_t)b * DD * SS, SS, (nt0 + j) * 16, jt * 32, lane);
                acc[j] = wmma_f16(a, bf, acc[j]);
            }
        }
        int n = lane & 15, hi = (lane >> 4) & 1;
#pragma unroll
        for (int j = 0; j < 3; ++j)
#pragma unroll
            for (int r = 0; r < 8; ++r) {
                int mrow = r + hi * 8;
                cl[(rowG + mrow) * DD + (nt0 + j) * 16 + n] = (_Float16)(acc[j][r] * rinv[mrow]);
            }
    }
}

// ---------------- kernel 6: output projection h = CL @ Wo + bo; 2x4 tiles per wave
__global__ void k_oproj(const _Float16* __restrict__ cl, const _Float16* __restrict__ Wot,
                        const float* __restrict__ bo, float* __restrict__ h) {
    int wid  = blockIdx.x * 8 + (threadIdx.x >> 5);   // 0..6143
    int lane = threadIdx.x & 31;
    int mg   = wid / 12;                              // 0..511
    int ntB  = (wid - mg * 12) * 4;                   // 0..44
    int rowBase = mg * 32;
    v8f acc[2][4] = {};
    for (int kk = 0; kk < DD; kk += 32) {
        v16h a0 = load_a_frag(cl, DD, rowBase,      kk, lane);
        v16h a1 = load_a_frag(cl, DD, rowBase + 16, kk, lane);
#pragma unroll
        for (int j = 0; j < 4; ++j) {
            v16h b = load_b_frag(Wot, DD, (ntB + j) * 16, kk, lane);
            acc[0][j] = wmma_f16(a0, b, acc[0][j]);
            acc[1][j] = wmma_f16(a1, b, acc[1][j]);
        }
    }
    int n = lane & 15, hi = (lane >> 4) & 1;
#pragma unroll
    for (int j = 0; j < 4; ++j) {
        int col = (ntB + j) * 16 + n;
        float bbv = bo[col];
#pragma unroll
        for (int i = 0; i < 2; ++i)
#pragma unroll
            for (int r = 0; r < 8; ++r)
                h[(size_t)(rowBase + i * 16 + r + hi * 8) * DD + col] = acc[i][j][r] + bbv;
    }
}

// ---------------- kernel 7: residual + LayerNorm (eps 1e-5) + actor head softmax
__global__ void k_final(const float* __restrict__ h, const float* __restrict__ ctx,
                        const float* __restrict__ g, const float* __restrict__ bt,
                        const float* __restrict__ Wa, const float* __restrict__ ba,
                        float* __restrict__ out) {
    __shared__ float red[256];
    __shared__ float L0s;
    int r = blockIdx.x, t = threadIdx.x;
    const float* hr = h + (size_t)r * DD;
    const float* cr = ctx + (size_t)r * DD;
    float x0 = hr[t] + cr[t], x1 = hr[t + 256] + cr[t + 256], x2 = hr[t + 512] + cr[t + 512];
    red[t] = x0 + x1 + x2; __syncthreads();
    for (int off = 128; off > 0; off >>= 1) { if (t < off) red[t] += red[t + off]; __syncthreads(); }
    float mean = red[0] * (1.0f / (float)DD); __syncthreads();
    float d0 = x0 - mean, d1 = x1 - mean, d2 = x2 - mean;
    red[t] = d0 * d0 + d1 * d1 + d2 * d2; __syncthreads();
    for (int off = 128; off > 0; off >>= 1) { if (t < off) red[t] += red[t + off]; __syncthreads(); }
    float rs = rsqrtf(red[0] * (1.0f / (float)DD) + 1e-5f); __syncthreads();
    float y0 = g[t] * d0 * rs + bt[t];
    float y1 = g[t + 256] * d1 * rs + bt[t + 256];
    float y2 = g[t + 512] * d2 * rs + bt[t + 512];
    float* emb = out + (size_t)BSR * 2;
    size_t base = (size_t)r * DD;
    emb[base + t] = y0; emb[base + t + 256] = y1; emb[base + t + 512] = y2;
    float l0 = y0 * Wa[t * 2] + y1 * Wa[(t + 256) * 2] + y2 * Wa[(t + 512) * 2];
    float l1 = y0 * Wa[t * 2 + 1] + y1 * Wa[(t + 256) * 2 + 1] + y2 * Wa[(t + 512) * 2 + 1];
    red[t] = l0; __syncthreads();
    for (int off = 128; off > 0; off >>= 1) { if (t < off) red[t] += red[t + off]; __syncthreads(); }
    if (t == 0) L0s = red[0] + ba[0];
    __syncthreads();
    red[t] = l1; __syncthreads();
    for (int off = 128; off > 0; off >>= 1) { if (t < off) red[t] += red[t + off]; __syncthreads(); }
    if (t == 0) {
        float La = L0s, Lb = red[0] + ba[1];
        float mm = fmaxf(La, Lb);
        float e0 = __expf(La - mm), e1 = __expf(Lb - mm);
        float inv = 1.0f / (e0 + e1);
        out[(size_t)r * 2]     = e0 * inv;
        out[(size_t)r * 2 + 1] = e1 * inv;
    }
}

extern "C" void kernel_launch(void* const* d_in, const int* in_sizes, int n_in,
                              void* d_out, int out_size, void* d_ws, size_t ws_size,
                              hipStream_t stream) {
    const float* context = (const float*)d_in[0];
    const int*   mask    = (const int*)d_in[1];
    const float* a2 = (const float*)d_in[2];
    const float* b2 = (const float*)d_in[3];
    const float* Wq = (const float*)d_in[4];
    const float* bq = (const float*)d_in[5];
    const float* Wk = (const float*)d_in[6];
    const float* bk = (const float*)d_in[7];
    const float* Wv = (const float*)d_in[8];
    const float* bv = (const float*)d_in[9];
    const float* Wo = (const float*)d_in[10];
    const float* bo = (const float*)d_in[11];
    const float* ln_g = (const float*)d_in[12];
    const float* ln_b = (const float*)d_in[13];
    const float* Wa = (const float*)d_in[14];
    const float* ba = (const float*)d_in[15];
    float* out = (float*)d_out;

    uintptr_t p = (uintptr_t)d_ws;
    auto carve = [&](size_t bytes) -> void* {
        void* q = (void*)p;
        p += (bytes + 255) & ~(size_t)255;
        return q;
    };
    float*    ctx  = (float*)   carve((size_t)BSR * DD * 4);
    _Float16* ctxh = (_Float16*)carve((size_t)BSR * DD * 2);
    _Float16* qh   = (_Float16*)carve((size_t)BSR * DD * 2);
    _Float16* kh   = (_Float16*)carve((size_t)BSR * DD * 2);
    _Float16* vT   = (_Float16*)carve((size_t)BSR * DD * 2);
    _Float16* cl   = (_Float16*)carve((size_t)BSR * DD * 2);
    float*    hbuf = (float*)   carve((size_t)BSR * DD * 4);
    _Float16* Wqt  = (_Float16*)carve((size_t)DD * DD * 2);
    _Float16* Wkt  = (_Float16*)carve((size_t)DD * DD * 2);
    _Float16* Wvt  = (_Float16*)carve((size_t)DD * DD * 2);
    _Float16* Wot  = (_Float16*)carve((size_t)DD * DD * 2);
    int*      len  = (int*)     carve(BB * sizeof(int));

    int wgrid = (DD * DD + 255) / 256;
    k_cvt_w<<<wgrid, 256, 0, stream>>>(Wq, Wqt);
    k_cvt_w<<<wgrid, 256, 0, stream>>>(Wk, Wkt);
    k_cvt_w<<<wgrid, 256, 0, stream>>>(Wv, Wvt);
    k_cvt_w<<<wgrid, 256, 0, stream>>>(Wo, Wot);

    k_lengths<<<BB, 256, 0, stream>>>(mask, len);
    k_norm<<<BSR, 256, 0, stream>>>(context, a2, b2, ctx, ctxh);
    // 512 M-groups * 36 (3 mats * 12 col-groups) wave-jobs, 8 waves/block
    k_qkv<<<512 * 36 / 8, 256, 0, stream>>>(ctxh, Wqt, Wkt, Wvt, bq, bk, bv, qh, kh, vT);
    k_attn<<<BSR / 16, 256, 0, stream>>>(qh, kh, vT, mask, len, cl);
    // 512 M-groups * 12 col-groups, 8 waves/block
    k_oproj<<<512 * 12 / 8, 256, 0, stream>>>(cl, Wot, bo, hbuf);
    k_final<<<BSR, 256, 0, stream>>>(hbuf, ctx, ln_g, ln_b, Wa, ba, out);
}
